// SchemaAggregator_32641751450077
// MI455X (gfx1250) — compile-verified, matrix operand order checked by reading, not run on hardware
//
#include <hip/hip_runtime.h>

typedef __attribute__((ext_vector_type(2)))  float  v2f;
typedef __attribute__((ext_vector_type(4)))  float  v4f;
typedef __attribute__((ext_vector_type(8)))  float  v8f;
typedef __attribute__((ext_vector_type(16))) __bf16 v16bf;

#if __has_builtin(__builtin_amdgcn_wmma_f32_16x16x4_f32)
#define SA_USE_WMMA_F32 1
#else
#define SA_USE_WMMA_F32 0
#endif

// ---------------------------------------------------------------------------
// Kernel 0: zero the workspace (S accumulator + counts). d_ws is poisoned.
// ---------------------------------------------------------------------------
__global__ void sa_zero_kernel(float* __restrict__ p, int n) {
    int i = blockIdx.x * blockDim.x + threadIdx.x;
    int stride = gridDim.x * blockDim.x;
    for (; i < n; i += stride) p[i] = 0.0f;
}

// ---------------------------------------------------------------------------
// Kernel 1: streaming segment-sum of x into S[B,H] + per-segment row counts.
// seg_ids is sorted -> each block owns a contiguous row range, accumulates in
// registers per run, and only touches global atomics at segment transitions.
// 64 threads/block, each thread owns 4 consecutive columns (float4 loads).
// This kernel is the entire runtime: 512 MB / 23.3 TB/s ~= 22 us.
// ---------------------------------------------------------------------------
__global__ void sa_segsum_kernel(const float* __restrict__ x,
                                 const int*   __restrict__ seg,
                                 float* __restrict__ S,
                                 float* __restrict__ counts,
                                 int N, int H, int rows_per_block) {
    const int c4 = threadIdx.x * 4;              // column base for this thread
    const int r0 = blockIdx.x * rows_per_block;
    int r1 = r0 + rows_per_block;
    if (r1 > N) r1 = N;
    if (r0 >= r1) return;

    const float* xp = x + (size_t)r0 * H + c4;
    v4f acc = {0.f, 0.f, 0.f, 0.f};
    float cnt = 0.f;
    int cur = seg[r0];

    for (int r = r0; r < r1; ++r) {
        int s = seg[r];
        if (s != cur) {
            float* dst = S + (size_t)cur * H + c4;
            atomicAdd(dst + 0, acc.x);
            atomicAdd(dst + 1, acc.y);
            atomicAdd(dst + 2, acc.z);
            atomicAdd(dst + 3, acc.w);
            if (threadIdx.x == 0) atomicAdd(&counts[cur], cnt);
            acc = (v4f){0.f, 0.f, 0.f, 0.f};
            cnt = 0.f;
            cur = s;
        }
        // prefetch a few rows ahead of the HBM stream (x >> L2 capacity)
        __builtin_prefetch(xp + (size_t)8 * H, 0, 0);
        v4f v = *(const v4f*)xp;                 // global_load_b128
        acc.x += v.x; acc.y += v.y; acc.z += v.z; acc.w += v.w;
        cnt += 1.f;
        xp += H;
    }
    float* dst = S + (size_t)cur * H + c4;
    atomicAdd(dst + 0, acc.x);
    atomicAdd(dst + 1, acc.y);
    atomicAdd(dst + 2, acc.z);
    atomicAdd(dst + 3, acc.w);
    if (threadIdx.x == 0) atomicAdd(&counts[cur], cnt);
}

// ---------------------------------------------------------------------------
// Kernel 2: out[B,H] = relu(S @ W^T + counts (x) bias) via WMMA.
// One wave per 16x16 output tile. B-matrix element (k,n) = W[n][k], so both A
// and B fragments are contiguous little row slices -> global_load_b64.
// fp32 path: V_WMMA_F32_16X16X4_F32 (exact precision vs reference).
//   A 16x4 layout: lanes 0-15 hold K={0,1}, lanes 16-31 hold K={2,3}.
// Fallback: V_WMMA_F32_16X16X32_BF16 with ISA 7.12.2 16-bit A/B layout.
// ---------------------------------------------------------------------------
__global__ void sa_gemm_wmma_kernel(const float* __restrict__ S,
                                    const float* __restrict__ W,
                                    const float* __restrict__ bias,
                                    const float* __restrict__ counts,
                                    float* __restrict__ out,
                                    int H, int n_tiles) {
    const int lane = threadIdx.x & 31;
    const int wave = threadIdx.x >> 5;
    const int tile = blockIdx.x * (blockDim.x >> 5) + wave;
    if (tile >= n_tiles) return;                 // wave-uniform: EXEC stays all-1s

    const int ntn = H >> 4;                      // tiles along N (=16 for H=256)
    const int m0 = (tile / ntn) * 16;
    const int n0 = (tile % ntn) * 16;
    const int nn = lane & 15;
    const int hi = lane >> 4;

    const float* Arow = S + (size_t)(m0 + nn) * H;   // S row for this lane's M
    const float* Brow = W + (size_t)(n0 + nn) * H;   // W row for this lane's N

    v8f c = {};
#if SA_USE_WMMA_F32
    for (int k0 = 0; k0 < H; k0 += 4) {
        const int kb = k0 + hi * 2;              // lanes 16-31 take K+2,K+3
        v2f a = *(const v2f*)(Arow + kb);        // global_load_b64
        v2f b = *(const v2f*)(Brow + kb);
        c = __builtin_amdgcn_wmma_f32_16x16x4_f32(
                false, a, false, b, (short)0, c, false, false);
    }
#else
    for (int k0 = 0; k0 < H; k0 += 32) {
        v16bf a, b;
#pragma unroll
        for (int i = 0; i < 16; ++i) {
            // ISA 16-bit A/B layout: K = 16*(i>>3) + 8*hi + (i&7)
            const int K = k0 + 16 * (i >> 3) + 8 * hi + (i & 7);
            a[i] = (__bf16)Arow[K];
            b[i] = (__bf16)Brow[K];
        }
        c = __builtin_amdgcn_wmma_f32_16x16x32_bf16(
                false, a, false, b, (short)0, c, false, false);
    }
#endif
    // C/D layout: VGPR v, lanes 0-15 -> M=v, lanes 16-31 -> M=v+8; N = lane&15.
    const float bj = bias[n0 + nn];
#pragma unroll
    for (int v = 0; v < 8; ++v) {
        const int row = m0 + v + hi * 8;
        float val = c[v] + counts[row] * bj;
        out[(size_t)row * H + (n0 + nn)] = val > 0.f ? val : 0.f;
    }
}

// ---------------------------------------------------------------------------
extern "C" void kernel_launch(void* const* d_in, const int* in_sizes, int n_in,
                              void* d_out, int out_size, void* d_ws, size_t ws_size,
                              hipStream_t stream) {
    const float* x    = (const float*)d_in[0];   // [N, H]
    const int*   seg  = (const int*)  d_in[1];   // [N] sorted
    const float* W    = (const float*)d_in[2];   // [H, H]
    const float* bias = (const float*)d_in[3];   // [H]

    const int N = in_sizes[1];                   // 500000
    const int H = in_sizes[3];                   // 256
    const int B = out_size / H;                  // 1024

    float* S      = (float*)d_ws;                // [B, H] accumulator
    float* counts = S + (size_t)B * H;           // [B]

    // 0) zero workspace
    const int zn = B * H + B;
    sa_zero_kernel<<<(zn + 255) / 256, 256, 0, stream>>>(S, zn);

    // 1) streaming segment-sum (the ~22us HBM-bound pass)
    const int ROWS_PER_BLOCK = 128;
    const int nblk = (N + ROWS_PER_BLOCK - 1) / ROWS_PER_BLOCK;
    sa_segsum_kernel<<<nblk, H / 4, 0, stream>>>(x, seg, S, counts, N, H,
                                                 ROWS_PER_BLOCK);

    // 2) tiny WMMA GEMM + bias + relu
    const int n_tiles = (B / 16) * (H / 16);     // 1024 wave-tiles
    const int waves_per_block = 8;               // 256 threads
    const int gblk = (n_tiles + waves_per_block - 1) / waves_per_block;
    sa_gemm_wmma_kernel<<<gblk, waves_per_block * 32, 0, stream>>>(
        S, W, bias, counts, (float*)d_out, H, n_tiles);
}
